// SubjectLayer_11862699672203
// MI455X (gfx1250) — matrix operand order repaired and checked by reading, not compiled, not run on hardware
//
#include <hip/hip_runtime.h>
#include <stdint.h>

// ---------------------------------------------------------------------------
// Subject-specific 1x1 conv (per-sample 256x256 channel-mix GEMM) for MI455X.
// bf16x2 split-precision WMMA: out = (Whi+Wlo) x (Xhi+Xlo), dropping lo*lo.
// Weights pre-converted once to bf16 hi/lo in d_ws, then staged to LDS per
// K-step via the Tensor Data Mover (tensor_load_to_lds + s_wait_tensorcnt),
// with hardware LDS row padding for conflict-free fragment reads.
// ---------------------------------------------------------------------------

typedef __attribute__((ext_vector_type(16))) __bf16        v16bf;
typedef __attribute__((ext_vector_type(8)))  float         v8f;
typedef __attribute__((ext_vector_type(8)))  unsigned int  v8u;
typedef __attribute__((ext_vector_type(4)))  unsigned int  u32x4;
typedef __attribute__((ext_vector_type(8)))  int           i32x8;
typedef __attribute__((ext_vector_type(4)))  int           i32x4;

#define C_DIM   256      // channels (M and K)
#define W_DIM   3000     // spatial width (N)
#define KSTEP   32       // K per WMMA stage
#define NTILE   64       // spatial columns per workgroup
#define LDA     34       // padded LDS row (halves) -> 17-bank stride, conflict-free
#define LDB     34
#define NKSTEPS (C_DIM / KSTEP)

// Pack the top 16 bits (truncated bf16) of two f32 bit patterns into one u32:
// result = { hi16(u1), hi16(u0) }  via a single v_perm_b32.
static __device__ __forceinline__ unsigned int pack_bf16_trunc(unsigned int u0,
                                                               unsigned int u1) {
    return __builtin_amdgcn_perm(u1, u0, 0x07060302u);
}
static __device__ __forceinline__ float trunc_hi_f32(float f) {
    unsigned int u = __builtin_bit_cast(unsigned int, f) & 0xFFFF0000u;
    return __builtin_bit_cast(float, u);
}

// TDM: 2D tile load (bf16 elements) global -> LDS with hardware row padding.
//   tensor: 256 x 256 bf16 (one subject's weight plane, hi or lo)
//   tile:   tile_dim1=256 rows x tile_dim0=32 halves (64B = 16 DWORDs) per row
//   pad:    +1 DWORD after every 16 DWORDs -> LDS row stride = 34 halves (LDA)
static __device__ __forceinline__ void tdm_load_w_slab(unsigned int lds_addr,
                                                       unsigned long long gaddr) {
    u32x4 g0;
    g0[0] = 1u;                                        // count=1 (valid), user mode
    g0[1] = lds_addr;                                  // LDS byte address
    g0[2] = (unsigned int)(gaddr & 0xFFFFFFFFull);     // global_addr[31:0]
    g0[3] = (unsigned int)(gaddr >> 32) | (2u << 30);  // global_addr[56:32] | type=2

    i32x8 g1;
    g1[0] = (int)((1u << 16)      // data_size = 2 bytes
                | (1u << 20)      // pad_enable
                | (3u << 22));    // pad_interval code 3 = 16 DWORDs
                                  // pad_amount code 0 = 1 DWORD
    g1[1] = (int)(256u << 16);    // tensor_dim0[15:0] = 256   (bits 63:48)
    g1[2] = (int)(256u << 16);    // tensor_dim1[15:0] = 256   (bits 95:80)
    g1[3] = (int)(32u  << 16);    // tile_dim0 = 32            (bits 127:112)
    g1[4] = (int)256;             // tile_dim1 = 256           (bits 143:128)
    g1[5] = (int)256;             // tensor_dim0_stride = 256  (bits 191:160)
    g1[6] = 0;
    g1[7] = 0;

    i32x4 z4 = {0, 0, 0, 0};                 // groups 2/3 unused (2D tensor)
    i32x8 z8 = {0, 0, 0, 0, 0, 0, 0, 0};     // extra group (6-arg toolchain form)
    __builtin_amdgcn_tensor_load_to_lds(g0, g1, z4, z4, z8, 0);
}

// ---------------------------------------------------------------------------
// Pre-pass: weight fp32 -> packed bf16 hi/lo planes in workspace.
// Element order preserved ([s][m][k], k fastest) so TDM can copy rows linearly.
// ---------------------------------------------------------------------------
__global__ __launch_bounds__(256)
void preconvert_weights_bf16x2(const float* __restrict__ weight,
                               unsigned int* __restrict__ whi,   // 2 bf16 / u32
                               unsigned int* __restrict__ wlo,
                               int nElem) {
    const int i4 = blockIdx.x * 256 + threadIdx.x;   // float4 index
    const int base = i4 << 2;
    if (base < nElem) {
        const float4 f4 = *(const float4*)(weight + base);
        const unsigned int u0 = __builtin_bit_cast(unsigned int, f4.x);
        const unsigned int u1 = __builtin_bit_cast(unsigned int, f4.y);
        const unsigned int u2 = __builtin_bit_cast(unsigned int, f4.z);
        const unsigned int u3 = __builtin_bit_cast(unsigned int, f4.w);
        const float r0 = f4.x - trunc_hi_f32(f4.x);
        const float r1 = f4.y - trunc_hi_f32(f4.y);
        const float r2 = f4.z - trunc_hi_f32(f4.z);
        const float r3 = f4.w - trunc_hi_f32(f4.w);
        uint2 h, l;
        h.x = pack_bf16_trunc(u0, u1);
        h.y = pack_bf16_trunc(u2, u3);
        l.x = pack_bf16_trunc(__builtin_bit_cast(unsigned int, r0),
                              __builtin_bit_cast(unsigned int, r1));
        l.y = pack_bf16_trunc(__builtin_bit_cast(unsigned int, r2),
                              __builtin_bit_cast(unsigned int, r3));
        *(uint2*)(whi + (size_t)i4 * 2) = h;
        *(uint2*)(wlo + (size_t)i4 * 2) = l;
    }
}

// ---------------------------------------------------------------------------
// Main GEMM kernel. PRECONV=true: A slab staged by TDM from bf16 workspace.
// PRECONV=false: self-contained (fp32 weight + in-kernel split), for small ws.
// ---------------------------------------------------------------------------
template <bool PRECONV>
__global__ __launch_bounds__(256)
void subject_conv1x1_wmma(const float*  __restrict__ x,       // [B, C, 1, W]
                          const float*  __restrict__ weight,  // [S, C, C] fp32
                          const __bf16* __restrict__ wHiG,    // [S, C, C] bf16 hi
                          const __bf16* __restrict__ wLoG,    // [S, C, C] bf16 lo
                          const float*  __restrict__ bias,    // [S, C]
                          const int*    __restrict__ subj,    // [B]
                          float*        __restrict__ out)     // [B, C, W]
{
    __shared__ __attribute__((aligned(16))) __bf16 a_hi[C_DIM * LDA];  // W slab [m][k]
    __shared__ __attribute__((aligned(16))) __bf16 a_lo[C_DIM * LDA];
    __shared__ __attribute__((aligned(16))) __bf16 b_hi[NTILE * LDB]; // X slab [n][k]
    __shared__ __attribute__((aligned(16))) __bf16 b_lo[NTILE * LDB];
    __shared__ float bias_s[C_DIM];

    const int tid  = threadIdx.x;
    const int b    = blockIdx.y;
    const int n0   = blockIdx.x * NTILE;
    const int s    = subj[b];

    const float* wS   = weight + (size_t)s * C_DIM * C_DIM;
    const float* xB   = x      + (size_t)b * C_DIM * W_DIM;
    float*       outB = out    + (size_t)b * C_DIM * W_DIM;

    bias_s[tid] = bias[s * C_DIM + tid];

    const int lane     = tid & 31;
    const int wv       = tid >> 5;
    const int mi       = wv >> 1;        // 0..3 : M quadrant of 64 rows
    const int ni       = wv & 1;         // 0..1 : N half of 32 cols
    const int lane15   = lane & 15;
    const int laneHalf = lane >> 4;      // 0 or 1

    // LDS byte addresses (flat pointer truncated to its low 32 bits).
    const unsigned int aHiLds = (unsigned int)(uintptr_t)(void*)a_hi;
    const unsigned int aLoLds = (unsigned int)(uintptr_t)(void*)a_lo;

    v8f acc[4][2];
#pragma unroll
    for (int mt = 0; mt < 4; ++mt)
#pragma unroll
        for (int nt = 0; nt < 2; ++nt)
#pragma unroll
            for (int v = 0; v < 8; ++v)
                acc[mt][nt][v] = 0.0f;

    for (int ks = 0; ks < NKSTEPS; ++ks) {
        const int k0 = ks * KSTEP;

        __syncthreads();   // previous stage's fragments fully consumed

        if constexpr (PRECONV) {
            // ---- Stage A via Tensor Data Mover (wave 0 issues; EXEC ignored) ----
            if (wv == 0) {
                const size_t off = (size_t)s * C_DIM * C_DIM + k0;
                tdm_load_w_slab(aHiLds, (unsigned long long)(uintptr_t)(wHiG + off));
                tdm_load_w_slab(aLoLds, (unsigned long long)(uintptr_t)(wLoG + off));
            }
        } else {
            // ---- Stage A: fp32 -> bf16 hi/lo via perm packing (fallback) ----
#pragma unroll
            for (int i = 0; i < 8; ++i) {
                const int l = i * 256 + tid;          // 2048 float4 chunks
                const int m = l >> 3;
                const int q = (l & 7) << 2;
                const float4 f4 = *(const float4*)(wS + (size_t)m * C_DIM + k0 + q);
                const unsigned int u0 = __builtin_bit_cast(unsigned int, f4.x);
                const unsigned int u1 = __builtin_bit_cast(unsigned int, f4.y);
                const unsigned int u2 = __builtin_bit_cast(unsigned int, f4.z);
                const unsigned int u3 = __builtin_bit_cast(unsigned int, f4.w);
                const float r0 = f4.x - trunc_hi_f32(f4.x);
                const float r1 = f4.y - trunc_hi_f32(f4.y);
                const float r2 = f4.z - trunc_hi_f32(f4.z);
                const float r3 = f4.w - trunc_hi_f32(f4.w);
                unsigned int* ph = (unsigned int*)&a_hi[m * LDA + q];
                unsigned int* pl = (unsigned int*)&a_lo[m * LDA + q];
                ph[0] = pack_bf16_trunc(u0, u1);
                ph[1] = pack_bf16_trunc(u2, u3);
                pl[0] = pack_bf16_trunc(__builtin_bit_cast(unsigned int, r0),
                                        __builtin_bit_cast(unsigned int, r1));
                pl[1] = pack_bf16_trunc(__builtin_bit_cast(unsigned int, r2),
                                        __builtin_bit_cast(unsigned int, r3));
            }
        }

        // ---- Stage B: X[b][k0..k0+31][n0..n0+63] (zero-pad the 3000 tail) ----
#pragma unroll
        for (int i = 0; i < 8; ++i) {
            const int l   = i * 256 + tid;        // 2048 scalars, n fastest (coalesced)
            const int k   = l >> 6;
            const int n   = l & 63;
            const int col = n0 + n;
            const float f = (col < W_DIM) ? xB[(size_t)(k0 + k) * W_DIM + col] : 0.0f;
            const unsigned int uf = __builtin_bit_cast(unsigned int, f);
            const float        r  = f - trunc_hi_f32(f);
            const unsigned int ur = __builtin_bit_cast(unsigned int, r);
            b_hi[n * LDB + k] = __builtin_bit_cast(__bf16, (unsigned short)(uf >> 16));
            b_lo[n * LDB + k] = __builtin_bit_cast(__bf16, (unsigned short)(ur >> 16));
        }

        // Prefetch next x K-slab while DMA/compute proceed (global_prefetch_b8).
        if (ks + 1 < NKSTEPS) {
            __builtin_prefetch(xB + (size_t)(k0 + KSTEP + (tid >> 3)) * W_DIM
                                  + n0 + ((tid & 7) << 3), 0, 1);
            if constexpr (!PRECONV)
                __builtin_prefetch(wS + (size_t)tid * C_DIM + k0 + KSTEP, 0, 1);
        }

        if constexpr (PRECONV)
            __builtin_amdgcn_s_wait_tensorcnt(0);   // no-op for waves with cnt==0

        __syncthreads();

        // ---- Build B fragments (ISA: lanes 0-15 K=0..7,16..23, lanes 16-31 K=8..15,24..31) ----
        v16bf bhf[2], blf[2];
#pragma unroll
        for (int nt = 0; nt < 2; ++nt) {
            const int n  = ni * 32 + nt * 16 + lane15;
            const int kb = laneHalf * 8;
            const unsigned int* ph = (const unsigned int*)&b_hi[n * LDB + kb];
            const unsigned int* pl = (const unsigned int*)&b_lo[n * LDB + kb];
            v8u th, tl;
#pragma unroll
            for (int p = 0; p < 8; ++p) { th[p] = ph[(p & 3) + ((p >> 2) << 3)]; tl[p] = pl[(p & 3) + ((p >> 2) << 3)]; }
            bhf[nt] = __builtin_bit_cast(v16bf, th);
            blf[nt] = __builtin_bit_cast(v16bf, tl);
        }

        // ---- Build A fragments (ISA 16-bit A 16x32 layout) ----
        v16bf ahf[4], alf[4];
#pragma unroll
        for (int mt = 0; mt < 4; ++mt) {
            const int r  = mi * 64 + mt * 16 + lane15;
            const int kb = laneHalf * 8;
            v8u th, tl;
#pragma unroll
            for (int p = 0; p < 8; ++p) {
                const int kk = kb + ((p & 3) << 1) + ((p >> 2) << 4);
                th[p] = *(const unsigned int*)&a_hi[r * LDA + kk];
                tl[p] = *(const unsigned int*)&a_lo[r * LDA + kk];
            }
            ahf[mt] = __builtin_bit_cast(v16bf, th);
            alf[mt] = __builtin_bit_cast(v16bf, tl);
        }

        // ---- 3-product bf16x2 WMMA: hi*hi + hi*lo + lo*hi ----
#pragma unroll
        for (int mt = 0; mt < 4; ++mt) {
#pragma unroll
            for (int nt = 0; nt < 2; ++nt) {
                v8f c = acc[mt][nt];
                c = __builtin_amdgcn_wmma_f32_16x16x32_bf16(false, ahf[mt], false, bhf[nt],
                                                            (short)0, c, false, false);
                c = __builtin_amdgcn_wmma_f32_16x16x32_bf16(false, ahf[mt], false, blf[nt],
                                                            (short)0, c, false, false);
                c = __builtin_amdgcn_wmma_f32_16x16x32_bf16(false, alf[mt], false, bhf[nt],
                                                            (short)0, c, false, false);
                acc[mt][nt] = c;
            }
        }
    }

    // ---- Store: C/D layout -> row = base + 8*(lane>=16) + vgpr, col = lane&15 ----
#pragma unroll
    for (int mt = 0; mt < 4; ++mt) {
#pragma unroll
        for (int nt = 0; nt < 2; ++nt) {
            const int col = n0 + ni * 32 + nt * 16 + lane15;
            if (col < W_DIM) {
                const int rowBase = mi * 64 + mt * 16 + laneHalf * 8;
#pragma unroll
                for (int v = 0; v < 8; ++v) {
                    const int row = rowBase + v;
                    outB[(size_t)row * W_DIM + col] = acc[mt][nt][v] + bias_s[row];
                }
            }
        }
    }
}

extern "C" void kernel_launch(void* const* d_in, const int* in_sizes, int n_in,
                              void* d_out, int out_size, void* d_ws, size_t ws_size,
                              hipStream_t stream) {
    const float* x      = (const float*)d_in[0];
    const float* weight = (const float*)d_in[1];
    const float* bias   = (const float*)d_in[2];
    const int*   subj   = (const int*)d_in[3];
    float*       out    = (float*)d_out;

    const int Bn = in_sizes[3];                       // batch = #subject indices
    const size_t wElems = (size_t)in_sizes[1];        // S*C*C
    const size_t needWs = wElems * 4;                 // bf16 hi + bf16 lo planes

    dim3 grid((W_DIM + NTILE - 1) / NTILE, Bn);

    if (ws_size >= needWs) {
        unsigned int* whi = (unsigned int*)d_ws;
        unsigned int* wlo = (unsigned int*)((char*)d_ws + wElems * sizeof(__bf16));
        const int nPre = (int)((wElems / 4 + 255) / 256);
        hipLaunchKernelGGL(preconvert_weights_bf16x2, dim3(nPre), dim3(256), 0, stream,
                           weight, whi, wlo, (int)wElems);
        hipLaunchKernelGGL((subject_conv1x1_wmma<true>), grid, dim3(256), 0, stream,
                           x, weight, (const __bf16*)whi, (const __bf16*)wlo,
                           bias, subj, out);
    } else {
        hipLaunchKernelGGL((subject_conv1x1_wmma<false>), grid, dim3(256), 0, stream,
                           x, weight, (const __bf16*)nullptr, (const __bf16*)nullptr,
                           bias, subj, out);
    }
}
